// GATLinkPredictor_37563783970933
// MI455X (gfx1250) — matrix operand order repaired and checked by reading, not compiled, not run on hardware
//
#include <hip/hip_runtime.h>

typedef __attribute__((ext_vector_type(16))) _Float16 v16h;
typedef __attribute__((ext_vector_type(8)))  _Float16 half8v;
typedef __attribute__((ext_vector_type(4)))  _Float16 half4v;
typedef __attribute__((ext_vector_type(8)))  float    v8f;
typedef __attribute__((ext_vector_type(4)))  float    float4v;
typedef __attribute__((ext_vector_type(2)))  float    float2v;

// ---------- helpers ----------

// Order-preserving float -> uint mapping so segment-max can use u32 atomicMax.
__device__ __forceinline__ unsigned f2ord(float f) {
  unsigned u = __float_as_uint(f);
  return (u & 0x80000000u) ? ~u : (u | 0x80000000u);
}
__device__ __forceinline__ float ord2f(unsigned k) {
  unsigned u = (k & 0x80000000u) ? (k & 0x7fffffffu) : ~k;
  return __uint_as_float(u);
}

__device__ __forceinline__ float waveSum(float v) {
#pragma unroll
  for (int o = 16; o > 0; o >>= 1) v += __shfl_xor(v, o, 32);
  return v;
}

// ---------- WMMA GEMM:  Y[M,N] = (reluIn? relu(X) : X)[M,K] * W[K,N] + bias[N] ----------
// Block = 128 threads = 4 wave32.  Block tile 32(M) x 64(N); wave w owns a 16x32
// strip: tm=(w&1)*16, tn=(w>>1)*32 -> two v_wmma_f32_16x16x32_f16 sharing the A frag.
// Staging: float4 global loads -> f16 -> half4 LDS stores.  A rows 80B, B rows 160B
// (16B aligned for ds_load_b128 fragment reads).
#define ASTR 40   // halfs per A row (80 B)
#define BSTR 80   // halfs per B row (160 B)

__global__ __launch_bounds__(128)
void gemm_bias_wmma(const float* __restrict__ X, const float* __restrict__ W,
                    const float* __restrict__ bias, float* __restrict__ Y,
                    int M, int K, int N, int reluIn)
{
  __shared__ _Float16 As[32 * ASTR];   // 2.5 KB
  __shared__ _Float16 Bs[32 * BSTR];   // 5.0 KB

  const int tid  = threadIdx.x;
  const int lane = tid & 31;
  const int w    = tid >> 5;
  const int m0   = blockIdx.x * 32;
  const int n0   = blockIdx.y * 64;
  const int tm   = (w & 1) * 16;
  const int tn   = (w >> 1) * 32;

  v8f c0 = {}, c1 = {};
  for (int k0 = 0; k0 < K; k0 += 32) {
    // stage A: 32x32 f32 -> f16 (2 float4 per thread)
#pragma unroll
    for (int i = 0; i < 2; ++i) {
      int chunk = tid + 128 * i;
      int r = chunk >> 3, cc = (chunk & 7) * 4;
      float4v xv = {0.f, 0.f, 0.f, 0.f};
      if (m0 + r < M) xv = *(const float4v*)(X + (m0 + r) * K + k0 + cc);
      half4v hv;
#pragma unroll
      for (int j = 0; j < 4; ++j) {
        float t = xv[j];
        if (reluIn) t = fmaxf(t, 0.f);
        hv[j] = (_Float16)t;
      }
      *(half4v*)&As[r * ASTR + cc] = hv;
    }
    // stage B: 32x64 f32 -> f16 (4 float4 per thread)
#pragma unroll
    for (int i = 0; i < 4; ++i) {
      int chunk = tid + 128 * i;
      int r = chunk >> 4, cc = (chunk & 15) * 4;
      float4v wv = *(const float4v*)(W + (k0 + r) * N + n0 + cc);
      half4v hv;
#pragma unroll
      for (int j = 0; j < 4; ++j) hv[j] = (_Float16)wv[j];
      *(half4v*)&Bs[r * BSTR + cc] = hv;
    }
    __syncthreads();

    // A fragment (ISA 7.12.2): lane m = tm+(lane&15); halves[0..7] = K hi..hi+7,
    // halves[8..15] = K 16+hi..23+hi, hi = 8*(lane>>4).  Two 16B chunks.
    const int mrow = tm + (lane & 15);
    const int hi   = (lane >> 4) * 8;
    const _Float16* ap = &As[mrow * ASTR + hi];
    half8v alo = *(const half8v*)ap;
    half8v ahi = *(const half8v*)(ap + 16);
    v16h a;
#pragma unroll
    for (int j = 0; j < 8; ++j) { a[j] = alo[j]; a[8 + j] = ahi[j]; }

    // B fragment: lane n = tn+(lane&15); halves[0..7] = K hi..hi+7, halves[8..15] = K 16+hi..23+hi
    v16h b0, b1;
    const int ncol = tn + (lane & 15);
#pragma unroll
    for (int j = 0; j < 8; ++j) {
      b0[j]     = Bs[(hi + j) * BSTR + ncol];
      b0[8 + j] = Bs[(16 + hi + j) * BSTR + ncol];
      b1[j]     = Bs[(hi + j) * BSTR + ncol + 16];
      b1[8 + j] = Bs[(16 + hi + j) * BSTR + ncol + 16];
    }

    c0 = __builtin_amdgcn_wmma_f32_16x16x32_f16(false, a, false, b0, (short)0, c0, false, false);
    c1 = __builtin_amdgcn_wmma_f32_16x16x32_f16(false, a, false, b1, (short)0, c1, false, false);
    __syncthreads();
  }

  // C/D layout: VGPR r, lane -> M = r + 8*(lane>>4), N = lane&15
  const int col0 = n0 + tn + (lane & 15);
  const float bv0 = bias[col0];
  const float bv1 = bias[col0 + 16];
#pragma unroll
  for (int r = 0; r < 8; ++r) {
    int row = m0 + tm + r + (lane >> 4) * 8;
    if (row < M) {
      Y[row * N + col0]      = c0[r] + bv0;
      Y[row * N + col0 + 16] = c1[r] + bv1;
    }
  }
}

// ---------- init: out[n][:] = bias, segment stats cleared ----------
__global__ void init_layer(float* __restrict__ out, const float* __restrict__ bias,
                           int n, int HC, unsigned* __restrict__ mkey,
                           float* __restrict__ denom, int nh)
{
  int total = n * HC;
  for (int id = blockIdx.x * blockDim.x + threadIdx.x; id < total;
       id += gridDim.x * blockDim.x) {
    out[id] = bias[id % HC];
    if (id < nh) { mkey[id] = 0u; denom[id] = 0.f; }
  }
}

// ---------- pass 1: per-edge attention logits + segment max ----------
// Wave per edge; lane owns VEC = HC/32 contiguous channels (VEC | C so a chunk
// never straddles a head).  Head sums via segmented shfl_xor over G = C/VEC lanes.
template<int HC, int H, int C>
__global__ __launch_bounds__(256)
void edge_score_max(const float* __restrict__ xl, const float* __restrict__ xr,
                    const int* __restrict__ esrc, const int* __restrict__ edst,
                    int E, int Etot, const float* __restrict__ att,
                    float* __restrict__ score, unsigned* __restrict__ mkey)
{
  const int VEC = HC / 32;
  const int G   = C / VEC;            // lanes per head group (32/H)
  const int lane = threadIdx.x & 31;
  const int base = lane * VEC;
  int wid = (blockIdx.x * blockDim.x + threadIdx.x) >> 5;
  const int nw = (gridDim.x * blockDim.x) >> 5;
  for (int e = wid; e < Etot; e += nw) {
    const int s = (e < E) ? esrc[e] : (e - E);   // self-loops appended
    const int d = (e < E) ? edst[e] : (e - E);
    float p = 0.f;
    if (VEC >= 4) {
#pragma unroll
      for (int v = 0; v < VEC / 4; ++v) {
        float4v xa = *(const float4v*)(xl + s * HC + base + v * 4);
        float4v xb = *(const float4v*)(xr + d * HC + base + v * 4);
        float4v av = *(const float4v*)(att + base + v * 4);
#pragma unroll
        for (int j = 0; j < 4; ++j) {
          float g = xa[j] + xb[j];
          g = (g > 0.f) ? g : 0.2f * g;          // LeakyReLU(0.2)
          p += g * av[j];
        }
      }
    } else {
      float2v xa = *(const float2v*)(xl + s * HC + base);
      float2v xb = *(const float2v*)(xr + d * HC + base);
      float2v av = *(const float2v*)(att + base);
#pragma unroll
      for (int j = 0; j < 2; ++j) {
        float g = xa[j] + xb[j];
        g = (g > 0.f) ? g : 0.2f * g;
        p += g * av[j];
      }
    }
#pragma unroll
    for (int o = G / 2; o > 0; o >>= 1) p += __shfl_xor(p, o, 32);
    if ((lane & (G - 1)) == 0) {
      int h = lane / G;
      score[e * H + h] = p;
      atomicMax(&mkey[d * H + h], f2ord(p));
    }
  }
}

// ---------- pass 2: exp(score - max), segment sum ----------
__global__ __launch_bounds__(256)
void edge_exp(const int* __restrict__ edst, int E, int Etot, int H,
              float* __restrict__ score, const unsigned* __restrict__ mkey,
              float* __restrict__ denom)
{
  int total = Etot * H;
  for (int id = blockIdx.x * blockDim.x + threadIdx.x; id < total;
       id += gridDim.x * blockDim.x) {
    int e = id / H, h = id - e * H;
    int d = (e < E) ? edst[e] : (e - E);
    float m  = ord2f(mkey[d * H + h]);
    float ex = __expf(score[id] - m);
    score[id] = ex;                               // overwrite in place
    atomicAdd(&denom[d * H + h], ex);
  }
}

// ---------- pass 3: out[d] += alpha * xl[s] ----------
template<int HC, int H, int C>
__global__ __launch_bounds__(256)
void edge_aggregate(const float* __restrict__ xl,
                    const int* __restrict__ esrc, const int* __restrict__ edst,
                    int E, int Etot, const float* __restrict__ ex,
                    const float* __restrict__ denom, float* __restrict__ out)
{
  const int VEC = HC / 32;
  const int G   = C / VEC;
  const int lane = threadIdx.x & 31;
  const int base = lane * VEC;
  const int h    = lane / G;                      // this lane's head
  int wid = (blockIdx.x * blockDim.x + threadIdx.x) >> 5;
  const int nw = (gridDim.x * blockDim.x) >> 5;
  for (int e = wid; e < Etot; e += nw) {
    const int s = (e < E) ? esrc[e] : (e - E);
    const int d = (e < E) ? edst[e] : (e - E);
    const float alpha = ex[e * H + h] / denom[d * H + h];
    if (VEC >= 4) {
#pragma unroll
      for (int v = 0; v < VEC / 4; ++v) {
        float4v xa = *(const float4v*)(xl + s * HC + base + v * 4);
#pragma unroll
        for (int j = 0; j < 4; ++j)
          atomicAdd(&out[d * HC + base + v * 4 + j], alpha * xa[j]);
      }
    } else {
      float2v xa = *(const float2v*)(xl + s * HC + base);
#pragma unroll
      for (int j = 0; j < 2; ++j)
        atomicAdd(&out[d * HC + base + j], alpha * xa[j]);
    }
  }
}

// ---------- fused link decoder: relu(concat(z[s],z[d]) @ Wm1 + bm1) @ Wm2 + bm2 ----------
__global__ __launch_bounds__(256)
void decoder(const float* __restrict__ z, const int* __restrict__ eli, int EL,
             const float* __restrict__ Wm1, const float* __restrict__ bm1,
             const float* __restrict__ Wm2, const float* __restrict__ bm2,
             float* __restrict__ out)
{
  __shared__ float sW[128 * 64];    // 32 KB: whole Wm1 in LDS
  __shared__ float sb[64];
  __shared__ float sw2[64];
  for (int i = threadIdx.x; i < 128 * 64; i += blockDim.x) sW[i] = Wm1[i];
  if (threadIdx.x < 64) { sb[threadIdx.x] = bm1[threadIdx.x]; sw2[threadIdx.x] = Wm2[threadIdx.x]; }
  __syncthreads();

  const int lane = threadIdx.x & 31;
  int wid = (blockIdx.x * blockDim.x + threadIdx.x) >> 5;
  const int nw = (gridDim.x * blockDim.x) >> 5;
  const float b2 = bm2[0];
  for (int e = wid; e < EL; e += nw) {
    const int s = eli[e];
    const int d = eli[EL + e];
    // coalesced register loads; broadcast inside the MAC loop via shfl
    float zs0 = z[s * 64 + lane], zs1 = z[s * 64 + lane + 32];
    float zd0 = z[d * 64 + lane], zd1 = z[d * 64 + lane + 32];
    float h0 = sb[lane], h1 = sb[lane + 32];   // lane owns hidden cols lane, lane+32
#pragma unroll
    for (int k = 0; k < 32; ++k) {
      float a = __shfl(zs0, k, 32);
      h0 += a * sW[k * 64 + lane];
      h1 += a * sW[k * 64 + lane + 32];
    }
#pragma unroll
    for (int k = 0; k < 32; ++k) {
      float a = __shfl(zs1, k, 32);
      h0 += a * sW[(k + 32) * 64 + lane];
      h1 += a * sW[(k + 32) * 64 + lane + 32];
    }
#pragma unroll
    for (int k = 0; k < 32; ++k) {
      float a = __shfl(zd0, k, 32);
      h0 += a * sW[(k + 64) * 64 + lane];
      h1 += a * sW[(k + 64) * 64 + lane + 32];
    }
#pragma unroll
    for (int k = 0; k < 32; ++k) {
      float a = __shfl(zd1, k, 32);
      h0 += a * sW[(k + 96) * 64 + lane];
      h1 += a * sW[(k + 96) * 64 + lane + 32];
    }
    h0 = fmaxf(h0, 0.f);
    h1 = fmaxf(h1, 0.f);
    float p = h0 * sw2[lane] + h1 * sw2[lane + 32];
    p = waveSum(p);
    if (lane == 0) out[e] = p + b2;
  }
}

// ---------- launcher ----------
extern "C" void kernel_launch(void* const* d_in, const int* in_sizes, int n_in,
                              void* d_out, int out_size, void* d_ws, size_t ws_size,
                              hipStream_t stream)
{
  const float* x     = (const float*)d_in[0];
  const int*   ei    = (const int*)  d_in[1];
  const int*   eli   = (const int*)  d_in[2];
  const float* W1l   = (const float*)d_in[3];
  const float* b1l   = (const float*)d_in[4];
  const float* W1r   = (const float*)d_in[5];
  const float* b1r   = (const float*)d_in[6];
  const float* att1  = (const float*)d_in[7];
  const float* bias1 = (const float*)d_in[8];
  const float* W2l   = (const float*)d_in[9];
  const float* b2l   = (const float*)d_in[10];
  const float* W2r   = (const float*)d_in[11];
  const float* b2r   = (const float*)d_in[12];
  const float* att2  = (const float*)d_in[13];
  const float* bias2 = (const float*)d_in[14];
  const float* Wm1   = (const float*)d_in[15];
  const float* bm1   = (const float*)d_in[16];
  const float* Wm2   = (const float*)d_in[17];
  const float* bm2   = (const float*)d_in[18];

  const int IN = 128, HID = 64, H1 = 4, OUT = 64;
  const int HC1 = H1 * HID;                       // 256
  const int N    = in_sizes[0] / IN;              // 50000
  const int E    = in_sizes[1] / 2;               // 800000
  const int EL   = in_sizes[2] / 2;               // 200000
  const int Etot = E + N;                         // + self loops

  // ---- workspace layout (with reuse; peak ~170 MB) ----
  char* ws = (char*)d_ws;
  size_t off = 0;
  auto alloc = [&](size_t bytes) -> void* {
    void* p = ws + off;
    off += (bytes + 255) & ~(size_t)255;
    return p;
  };
  float*    xl1 = (float*)alloc((size_t)N * HC1 * 4);
  float*    xr1 = (float*)alloc((size_t)N * HC1 * 4);
  float*    z1  = (float*)alloc((size_t)N * HC1 * 4);
  float*    sc1 = (float*)alloc((size_t)Etot * H1 * 4);
  unsigned* mk1 = (unsigned*)alloc((size_t)N * H1 * 4);
  float*    dn1 = (float*)alloc((size_t)N * H1 * 4);
  // layer-2 buffers reuse the (dead after their last use) xl1 / xr1 regions
  float*    xl2 = xl1;
  float*    xr2 = xl1 + (size_t)N * OUT;
  float*    z2  = xl1 + (size_t)2 * N * OUT;
  float*    sc2 = xr1;
  unsigned* mk2 = (unsigned*)(xr1 + (size_t)Etot);
  float*    dn2 = xr1 + (size_t)Etot + N;

  const int* esrc = ei;
  const int* edst = ei + E;

  const int WPB = 8;                              // waves per 256-thread block
  int eblocks = (Etot + WPB - 1) / WPB;

  // ===== layer 1 =====
  {
    dim3 grid((N + 31) / 32, HC1 / 64);
    gemm_bias_wmma<<<grid, 128, 0, stream>>>(x, W1l, b1l, xl1, N, IN, HC1, 0);
    gemm_bias_wmma<<<grid, 128, 0, stream>>>(x, W1r, b1r, xr1, N, IN, HC1, 0);
  }
  init_layer<<<(N * HC1 + 255) / 256, 256, 0, stream>>>(z1, bias1, N, HC1, mk1, dn1, N * H1);
  edge_score_max<256, 4, 64><<<eblocks, 256, 0, stream>>>(xl1, xr1, esrc, edst, E, Etot, att1, sc1, mk1);
  edge_exp<<<(Etot * H1 + 255) / 256, 256, 0, stream>>>(edst, E, Etot, H1, sc1, mk1, dn1);
  edge_aggregate<256, 4, 64><<<eblocks, 256, 0, stream>>>(xl1, esrc, edst, E, Etot, sc1, dn1, z1);

  // ===== layer 2 (ReLU fused into GEMM input load) =====
  {
    dim3 grid((N + 31) / 32, OUT / 64);
    gemm_bias_wmma<<<grid, 128, 0, stream>>>(z1, W2l, b2l, xl2, N, HC1, OUT, 1);
    gemm_bias_wmma<<<grid, 128, 0, stream>>>(z1, W2r, b2r, xr2, N, HC1, OUT, 1);
  }
  init_layer<<<(N * OUT + 255) / 256, 256, 0, stream>>>(z2, bias2, N, OUT, mk2, dn2, N);
  edge_score_max<64, 1, 64><<<eblocks, 256, 0, stream>>>(xl2, xr2, esrc, edst, E, Etot, att2, sc2, mk2);
  edge_exp<<<(Etot + 255) / 256, 256, 0, stream>>>(edst, E, Etot, 1, sc2, mk2, dn2);
  edge_aggregate<64, 1, 64><<<eblocks, 256, 0, stream>>>(xl2, esrc, edst, E, Etot, sc2, dn2, z2);

  // ===== decoder =====
  {
    int blocks = (EL + WPB - 1) / WPB;
    if (blocks > 8192) blocks = 8192;             // grid-stride; amortize LDS staging
    decoder<<<blocks, 256, 0, stream>>>(z2, eli, EL, Wm1, bm1, Wm2, bm2, (float*)d_out);
  }
}
